// BiMultiHeadAttention_8770323218993
// MI455X (gfx1250) — matrix operand order repaired and checked by reading, not compiled, not run on hardware
//
#include <hip/hip_runtime.h>

// ---------------------------------------------------------------------------
// BiMultiHeadAttention for MI455X (gfx1250, wave32, WMMA bf16 16x16x32)
// B=4, N=2048, E=256, H=8, D=32.
//  - All GEMM operands pre-packed bf16 so every fragment load is b128.
//  - Projection GEMMs: 32x64 register tile per wave (8 WMMA / 12 b128 loads).
//  - out_l: flash-style online softmax (single energy pass), emits m and 1/l
//    for the out_r = softmax^T @ V streaming pass.
// ---------------------------------------------------------------------------

typedef __attribute__((ext_vector_type(16))) __bf16 v16bf;
typedef __attribute__((ext_vector_type(8)))  float  v8f;

#define WMMA_BF16(A, B, C) \
    __builtin_amdgcn_wmma_f32_16x16x32_bf16(false, (A), false, (B), (short)0, (C), false, false)

#define CB 4
#define CN 2048
#define CE 256
#define CH 8
#define CD 32
#define CM (CB * CN)          // 8192 rows
#define CBH (CB * CH)         // 32
#define SCALE 0.17677669529663687f   // 1/sqrt(32)

union FragU { v16bf v; uint4 u[2]; };

// Fragment from two contiguous 8-element bf16 runs (16 B each).
__device__ __forceinline__ v16bf ld_frag(const __bf16* lo, const __bf16* hi) {
    FragU f;
    f.u[0] = *(const uint4*)lo;   // elements 0..7
    f.u[1] = *(const uint4*)hi;   // elements 8..15
    return f.v;
}
// Fragment from one contiguous 16-element bf16 run (32 B).
__device__ __forceinline__ v16bf ld_frag16(const __bf16* p) { return ld_frag(p, p + 8); }

// Convert 2x8 contiguous fp32 (LDS) to a bf16 fragment.
__device__ __forceinline__ v16bf cvt_frag(const float* lo, const float* hi) {
    v16bf r;
#pragma unroll
    for (int e = 0; e < 8; ++e) { r[e] = (__bf16)lo[e]; r[8 + e] = (__bf16)hi[e]; }
    return r;
}

// ---------------------------------------------------------------------------
// Pack x_l, x_r, (x_l - x_r) to bf16 once. 4 elems/thread, float4 in, uint2 out.
// ---------------------------------------------------------------------------
__global__ __launch_bounds__(256) void pack_x_kernel(
    const float* __restrict__ xl, const float* __restrict__ xr,
    __bf16* __restrict__ xq, __bf16* __restrict__ xk, __bf16* __restrict__ xv)
{
    const int idx = (blockIdx.x * 256 + threadIdx.x) * 4;
    const float4 a = *(const float4*)(xl + idx);
    const float4 b = *(const float4*)(xr + idx);
    union { __bf16 h[4]; uint2 u; } oq, ok, ov;
    oq.h[0] = (__bf16)a.x; oq.h[1] = (__bf16)a.y; oq.h[2] = (__bf16)a.z; oq.h[3] = (__bf16)a.w;
    ok.h[0] = (__bf16)b.x; ok.h[1] = (__bf16)b.y; ok.h[2] = (__bf16)b.z; ok.h[3] = (__bf16)b.w;
    ov.h[0] = (__bf16)(a.x - b.x); ov.h[1] = (__bf16)(a.y - b.y);
    ov.h[2] = (__bf16)(a.z - b.z); ov.h[3] = (__bf16)(a.w - b.w);
    *(uint2*)(xq + idx) = oq.u;
    *(uint2*)(xk + idx) = ok.u;
    *(uint2*)(xv + idx) = ov.u;
}

// ---------------------------------------------------------------------------
// Transpose + convert the four E x E weight matrices to bf16, k-contiguous:
// WT[n][k] = W[k][n].  grid (4, 16, 16), block (16, 16).
// ---------------------------------------------------------------------------
__global__ __launch_bounds__(256) void pack_wt_kernel(
    const float* __restrict__ W0, const float* __restrict__ W1,
    const float* __restrict__ W2, const float* __restrict__ W3,
    __bf16* __restrict__ T0, __bf16* __restrict__ T1,
    __bf16* __restrict__ T2, __bf16* __restrict__ T3)
{
    __shared__ float s[16][17];
    const float* W = (blockIdx.x == 0) ? W0 : (blockIdx.x == 1) ? W1
                   : (blockIdx.x == 2) ? W2 : W3;
    __bf16* T      = (blockIdx.x == 0) ? T0 : (blockIdx.x == 1) ? T1
                   : (blockIdx.x == 2) ? T2 : T3;
    const int kt = blockIdx.y, nt = blockIdx.z;
    const int tx = threadIdx.x, ty = threadIdx.y;
    s[ty][tx] = W[(kt * 16 + ty) * CE + nt * 16 + tx];
    __syncthreads();
    T[(nt * 16 + ty) * CE + kt * 16 + tx] = (__bf16)s[tx][ty];
}

// ---------------------------------------------------------------------------
// QKV GEMM: 32x64 register tile per wave; 8 WMMA per K-step off 12 b128 loads.
// Epilogue: bf16 out in [B,H,N,D]; optionally V^T [BH][D][N].
// grid (CM/64), block 256: 2 waves over M (64 rows), 4 waves over N (256 cols).
// ---------------------------------------------------------------------------
__global__ __launch_bounds__(256) void gemm_qkv_kernel(
    const __bf16* __restrict__ A, const __bf16* __restrict__ WT,
    const float* __restrict__ bias, __bf16* __restrict__ out,
    __bf16* __restrict__ vt)
{
    const int lane = threadIdx.x & 31;
    const int wv   = threadIdx.x >> 5;
    const int mw = wv >> 2, nw = wv & 3;
    const int mbase = blockIdx.x * 64 + mw * 32;
    const int nbase = nw * 64;
    const int g = lane >> 4, col = lane & 15;

    const __bf16* Ar0 = A + (size_t)(mbase + col) * CE;
    const __bf16* Ar1 = A + (size_t)(mbase + 16 + col) * CE;
    const __bf16* Bc0 = WT + (size_t)(nbase + col) * CE;
    const __bf16* Bc1 = WT + (size_t)(nbase + 16 + col) * CE;
    const __bf16* Bc2 = WT + (size_t)(nbase + 32 + col) * CE;
    const __bf16* Bc3 = WT + (size_t)(nbase + 48 + col) * CE;

    v8f acc[2][4] = {};
#pragma unroll
    for (int k0 = 0; k0 < CE; k0 += 32) {
        if (k0 + 32 < CE) {
            __builtin_prefetch(Ar0 + k0 + 32 + g * 8, 0, 1);
            __builtin_prefetch(Bc0 + k0 + 32 + g * 16, 0, 1);
        }
        const v16bf a0 = ld_frag(Ar0 + k0 + g * 16, Ar0 + k0 + g * 16 + 8);
        const v16bf a1 = ld_frag(Ar1 + k0 + g * 16, Ar1 + k0 + g * 16 + 8);
        const v16bf b0 = ld_frag16(Bc0 + k0 + g * 16);
        const v16bf b1 = ld_frag16(Bc1 + k0 + g * 16);
        const v16bf b2 = ld_frag16(Bc2 + k0 + g * 16);
        const v16bf b3 = ld_frag16(Bc3 + k0 + g * 16);
        acc[0][0] = WMMA_BF16(a0, b0, acc[0][0]);
        acc[0][1] = WMMA_BF16(a0, b1, acc[0][1]);
        acc[0][2] = WMMA_BF16(a0, b2, acc[0][2]);
        acc[0][3] = WMMA_BF16(a0, b3, acc[0][3]);
        acc[1][0] = WMMA_BF16(a1, b0, acc[1][0]);
        acc[1][1] = WMMA_BF16(a1, b1, acc[1][1]);
        acc[1][2] = WMMA_BF16(a1, b2, acc[1][2]);
        acc[1][3] = WMMA_BF16(a1, b3, acc[1][3]);
    }

#pragma unroll
    for (int j = 0; j < 4; ++j) {
        const int ncol = nbase + j * 16 + col;
        const float bs = bias[ncol];
        const int h = ncol >> 5, d = ncol & 31;
#pragma unroll
        for (int i = 0; i < 2; ++i) {
#pragma unroll
            for (int r = 0; r < 8; ++r) {
                const int m = mbase + i * 16 + r + g * 8;
                const int bidx = m >> 11, n = m & (CN - 1);
                const int bh = bidx * CH + h;
                const __bf16 val = (__bf16)(acc[i][j][r] + bs);
                out[((size_t)bh * CN + n) * CD + d] = val;
                if (vt) vt[((size_t)bh * CD + d) * CN + n] = val;
            }
        }
    }
}

// ---------------------------------------------------------------------------
// out_l with fused online softmax (flash style).  Wave owns a 16-row q tile,
// streams all key tiles once: energy WMMA -> running (m, l) + accumulator
// rescale -> P via per-wave LDS -> P@V WMMA.  Emits m and 1/l for out_r.
// grid (32, 16), block 256.
// ---------------------------------------------------------------------------
__global__ __launch_bounds__(256) void outl_flash_kernel(
    const __bf16* __restrict__ qb, const __bf16* __restrict__ kb,
    const __bf16* __restrict__ vt, float* __restrict__ mrow,
    float* __restrict__ linv, __bf16* __restrict__ outl)
{
    __shared__ float P[8][16][36];                // per-wave 16(q) x 32(kk)
    const int lane = threadIdx.x & 31;
    const int wv   = threadIdx.x >> 5;
    const int bh    = blockIdx.x;
    const int qtile = blockIdx.y * 8 + wv;
    const int g = lane >> 4, col = lane & 15;

    const __bf16* Q   = qb + (size_t)bh * CN * CD;
    const __bf16* K   = kb + (size_t)bh * CN * CD;
    const __bf16* VTb = vt + (size_t)bh * CD * CN;

    const __bf16* qp = Q + (qtile * 16 + col) * CD + g * 16;
    const v16bf qa = ld_frag(qp, qp + 8);

    float m8[8], l8[8];
#pragma unroll
    for (int r = 0; r < 8; ++r) { m8[r] = -1e30f; l8[r] = 0.0f; }

    v8f acc0 = {}, acc1 = {};
    for (int kt = 0; kt < CN / 16; kt += 2) {
        const v16bf kf0 = ld_frag16(K + (kt * 16 + col) * CD + g * 16);
        const v16bf kf1 = ld_frag16(K + ((kt + 1) * 16 + col) * CD + g * 16);
        v8f c0 = {}, c1 = {};
        c0 = WMMA_BF16(qa, kf0, c0);
        c1 = WMMA_BF16(qa, kf1, c1);

        float fr[8];
#pragma unroll
        for (int r = 0; r < 8; ++r) {
            const float v0 = c0[r] * SCALE, v1 = c1[r] * SCALE;
            float t = fmaxf(v0, v1);
#pragma unroll
            for (int off = 1; off < 16; off <<= 1)
                t = fmaxf(t, __shfl_xor(t, off, 32));       // row max (half-wave)
            const float mnew = fmaxf(m8[r], t);
            const float ef = __expf(m8[r] - mnew);
            const float p0 = __expf(v0 - mnew);
            const float p1 = __expf(v1 - mnew);
            float s = p0 + p1;
#pragma unroll
            for (int off = 1; off < 16; off <<= 1)
                s += __shfl_xor(s, off, 32);                // row sum (half-wave)
            l8[r] = l8[r] * ef + s;
            m8[r] = mnew;
            fr[r] = ef;
            P[wv][r + g * 8][col]      = p0;
            P[wv][r + g * 8][col + 16] = p1;
        }
#pragma unroll
        for (int r = 0; r < 8; ++r) { acc0[r] *= fr[r]; acc1[r] *= fr[r]; }

        const v16bf pa = cvt_frag(&P[wv][col][g * 16], &P[wv][col][g * 16 + 8]);
        const v16bf v0 = ld_frag16(VTb + (size_t)col * CN + kt * 16 + g * 16);
        const v16bf v1 = ld_frag16(VTb + (size_t)(col + 16) * CN + kt * 16 + g * 16);
        acc0 = WMMA_BF16(pa, v0, acc0);
        acc1 = WMMA_BF16(pa, v1, acc1);
    }

    float li[8];
#pragma unroll
    for (int r = 0; r < 8; ++r) li[r] = 1.0f / l8[r];

    const int h = bh & 7, b = bh >> 3;
#pragma unroll
    for (int r = 0; r < 8; ++r) {
        const int n = qtile * 16 + r + g * 8;
        outl[((size_t)(b * CN + n)) * CE + h * 32 + col]      = (__bf16)(acc0[r] * li[r]);
        outl[((size_t)(b * CN + n)) * CE + h * 32 + 16 + col] = (__bf16)(acc1[r] * li[r]);
    }
    if (col == 0) {
#pragma unroll
        for (int r = 0; r < 8; ++r) {
            const int row = bh * CN + qtile * 16 + r + g * 8;
            mrow[row] = m8[r];
            linv[row] = li[r];
        }
    }
}

// ---------------------------------------------------------------------------
// out_r = softmax(QK^T/s)^T @ V.  Wave owns a key tile, streams q tiles using
// m and 1/l from the flash pass; P stored transposed into per-wave LDS.
// ---------------------------------------------------------------------------
__global__ __launch_bounds__(256) void outr_kernel(
    const __bf16* __restrict__ qb, const __bf16* __restrict__ kb,
    const __bf16* __restrict__ vt, const float* __restrict__ mrow,
    const float* __restrict__ linv, __bf16* __restrict__ outr)
{
    __shared__ float PT[8][16][36];               // per-wave 16(kk) x 32(q)
    const int lane = threadIdx.x & 31;
    const int wv   = threadIdx.x >> 5;
    const int bh     = blockIdx.x;
    const int kktile = blockIdx.y * 8 + wv;
    const int g = lane >> 4, col = lane & 15;

    const __bf16* Q   = qb + (size_t)bh * CN * CD;
    const __bf16* K   = kb + (size_t)bh * CN * CD;
    const __bf16* VTb = vt + (size_t)bh * CD * CN;

    const v16bf kB = ld_frag16(K + (kktile * 16 + col) * CD + g * 16);

    v8f acc0 = {}, acc1 = {};
    for (int qt = 0; qt < CN / 16; qt += 2) {
#pragma unroll
        for (int s = 0; s < 2; ++s) {
            const __bf16* qp = Q + ((qt + s) * 16 + col) * CD + g * 16;
            const v16bf qa = ld_frag(qp, qp + 8);
            v8f c = {};
            c = WMMA_BF16(qa, kB, c);
#pragma unroll
            for (int r = 0; r < 8; ++r) {
                const int qrow = bh * CN + (qt + s) * 16 + r + g * 8;
                PT[wv][col][r + g * 8 + s * 16] =
                    __expf(c[r] * SCALE - mrow[qrow]) * linv[qrow];
            }
        }
        const v16bf pa = cvt_frag(&PT[wv][col][g * 16], &PT[wv][col][g * 16 + 8]);
        const v16bf v0 = ld_frag16(VTb + (size_t)col * CN + qt * 16 + g * 16);
        const v16bf v1 = ld_frag16(VTb + (size_t)(col + 16) * CN + qt * 16 + g * 16);
        acc0 = WMMA_BF16(pa, v0, acc0);
        acc1 = WMMA_BF16(pa, v1, acc1);
    }

    const int h = bh & 7, b = bh >> 3;
#pragma unroll
    for (int r = 0; r < 8; ++r) {
        const int n = kktile * 16 + r + g * 8;
        outr[((size_t)(b * CN + n)) * CE + h * 32 + col]      = (__bf16)acc0[r];
        outr[((size_t)(b * CN + n)) * CE + h * 32 + 16 + col] = (__bf16)acc1[r];
    }
}

// ---------------------------------------------------------------------------
// Output projection y = attn @ Wp + bp, 32x64 tile per wave, fp32 result.
// grid (CM/64), block 256.
// ---------------------------------------------------------------------------
__global__ __launch_bounds__(256) void gemm_proj_kernel(
    const __bf16* __restrict__ A, const __bf16* __restrict__ WT,
    const float* __restrict__ bias, float* __restrict__ y)
{
    const int lane = threadIdx.x & 31;
    const int wv   = threadIdx.x >> 5;
    const int mw = wv >> 2, nw = wv & 3;
    const int mbase = blockIdx.x * 64 + mw * 32;
    const int nbase = nw * 64;
    const int g = lane >> 4, col = lane & 15;

    const __bf16* Ar0 = A + (size_t)(mbase + col) * CE;
    const __bf16* Ar1 = A + (size_t)(mbase + 16 + col) * CE;
    const __bf16* Bc0 = WT + (size_t)(nbase + col) * CE;
    const __bf16* Bc1 = WT + (size_t)(nbase + 16 + col) * CE;
    const __bf16* Bc2 = WT + (size_t)(nbase + 32 + col) * CE;
    const __bf16* Bc3 = WT + (size_t)(nbase + 48 + col) * CE;

    v8f acc[2][4] = {};
#pragma unroll
    for (int k0 = 0; k0 < CE; k0 += 32) {
        if (k0 + 32 < CE) {
            __builtin_prefetch(Ar0 + k0 + 32 + g * 8, 0, 1);
            __builtin_prefetch(Bc0 + k0 + 32 + g * 16, 0, 1);
        }
        const v16bf a0 = ld_frag(Ar0 + k0 + g * 16, Ar0 + k0 + g * 16 + 8);
        const v16bf a1 = ld_frag(Ar1 + k0 + g * 16, Ar1 + k0 + g * 16 + 8);
        const v16bf b0 = ld_frag16(Bc0 + k0 + g * 16);
        const v16bf b1 = ld_frag16(Bc1 + k0 + g * 16);
        const v16bf b2 = ld_frag16(Bc2 + k0 + g * 16);
        const v16bf b3 = ld_frag16(Bc3 + k0 + g * 16);
        acc[0][0] = WMMA_BF16(a0, b0, acc[0][0]);
        acc[0][1] = WMMA_BF16(a0, b1, acc[0][1]);
        acc[0][2] = WMMA_BF16(a0, b2, acc[0][2]);
        acc[0][3] = WMMA_BF16(a0, b3, acc[0][3]);
        acc[1][0] = WMMA_BF16(a1, b0, acc[1][0]);
        acc[1][1] = WMMA_BF16(a1, b1, acc[1][1]);
        acc[1][2] = WMMA_BF16(a1, b2, acc[1][2]);
        acc[1][3] = WMMA_BF16(a1, b3, acc[1][3]);
    }

#pragma unroll
    for (int j = 0; j < 4; ++j) {
        const int ncol = nbase + j * 16 + col;
        const float bs = bias[ncol];
#pragma unroll
        for (int i = 0; i < 2; ++i) {
#pragma unroll
            for (int r = 0; r < 8; ++r)
                y[(size_t)(mbase + i * 16 + r + g * 8) * CE + ncol] = acc[i][j][r] + bs;
        }
    }
}

// ---------------------------------------------------------------------------
// LayerNorm(y)*gamma + beta + x.  One block (256 thr) per row.
// ---------------------------------------------------------------------------
__global__ __launch_bounds__(256) void ln_res_kernel(
    const float* __restrict__ y, const float* __restrict__ x,
    const float* __restrict__ gamma, const float* __restrict__ beta,
    float* __restrict__ out)
{
    const int row = blockIdx.x;
    const int t = threadIdx.x;
    const int lane = t & 31, wv = t >> 5;

    const float v = y[(size_t)row * CE + t];
    float s = v, s2 = v * v;
#pragma unroll
    for (int off = 16; off > 0; off >>= 1) {
        s  += __shfl_xor(s,  off, 32);
        s2 += __shfl_xor(s2, off, 32);
    }
    __shared__ float sh1[8], sh2[8], mvv[2];
    if (lane == 0) { sh1[wv] = s; sh2[wv] = s2; }
    __syncthreads();
    if (t == 0) {
        float a = 0.0f, b2 = 0.0f;
#pragma unroll
        for (int i = 0; i < 8; ++i) { a += sh1[i]; b2 += sh2[i]; }
        const float mean = a * (1.0f / CE);
        mvv[0] = mean;
        mvv[1] = b2 * (1.0f / CE) - mean * mean;
    }
    __syncthreads();
    const float mean = mvv[0];
    const float rinv = rsqrtf(mvv[1] + 1e-5f);
    out[(size_t)row * CE + t] =
        (v - mean) * rinv * gamma[t] + beta[t] + x[(size_t)row * CE + t];
}

// ---------------------------------------------------------------------------
// Host launcher
// ---------------------------------------------------------------------------
extern "C" void kernel_launch(void* const* d_in, const int* in_sizes, int n_in,
                              void* d_out, int out_size, void* d_ws, size_t ws_size,
                              hipStream_t stream)
{
    (void)in_sizes; (void)n_in; (void)out_size; (void)ws_size;

    const float* xl   = (const float*)d_in[0];
    const float* xr   = (const float*)d_in[1];
    const float* Wq   = (const float*)d_in[2];
    const float* bq   = (const float*)d_in[3];
    const float* Wk   = (const float*)d_in[4];
    const float* bk   = (const float*)d_in[5];
    const float* Wv   = (const float*)d_in[6];
    const float* bv   = (const float*)d_in[7];
    const float* Wp   = (const float*)d_in[8];
    const float* bp   = (const float*)d_in[9];
    const float* ln_g = (const float*)d_in[10];
    const float* ln_b = (const float*)d_in[11];
    const float* rn_g = (const float*)d_in[12];
    const float* rn_b = (const float*)d_in[13];

    char* ws = (char*)d_ws;
    size_t off = 0;
    __bf16* xq    = (__bf16*)(ws + off); off += (size_t)CM * CE * 2;   // 4 MB
    __bf16* xk    = (__bf16*)(ws + off); off += (size_t)CM * CE * 2;
    __bf16* xv    = (__bf16*)(ws + off); off += (size_t)CM * CE * 2;
    __bf16* WTq   = (__bf16*)(ws + off); off += (size_t)CE * CE * 2;   // 128 KB
    __bf16* WTk   = (__bf16*)(ws + off); off += (size_t)CE * CE * 2;
    __bf16* WTv   = (__bf16*)(ws + off); off += (size_t)CE * CE * 2;
    __bf16* WTp   = (__bf16*)(ws + off); off += (size_t)CE * CE * 2;
    __bf16* qb    = (__bf16*)(ws + off); off += (size_t)CM * CE * 2;
    __bf16* kb    = (__bf16*)(ws + off); off += (size_t)CM * CE * 2;
    __bf16* vbuf  = (__bf16*)(ws + off); off += (size_t)CM * CE * 2;
    __bf16* vt    = (__bf16*)(ws + off); off += (size_t)CM * CE * 2;   // V^T [BH][D][N]
    float*  mrow  = (float*) (ws + off); off += (size_t)CBH * CN * 4;  // 256 KB
    float*  linv  = (float*) (ws + off); off += (size_t)CBH * CN * 4;
    __bf16* attnl = (__bf16*)(ws + off); off += (size_t)CM * CE * 2;
    __bf16* attnr = (__bf16*)(ws + off); off += (size_t)CM * CE * 2;
    float*  yl    = (float*) (ws + off); off += (size_t)CM * CE * 4;   // 8 MB
    float*  yr    = (float*) (ws + off); off += (size_t)CM * CE * 4;

    float* outF = (float*)d_out;

    pack_x_kernel<<<(CM * CE) / 1024, 256, 0, stream>>>(xl, xr, xq, xk, xv);
    pack_wt_kernel<<<dim3(4, 16, 16), dim3(16, 16), 0, stream>>>(
        Wq, Wk, Wv, Wp, WTq, WTk, WTv, WTp);

    gemm_qkv_kernel<<<CM / 64, 256, 0, stream>>>(xq, WTq, bq, qb, (__bf16*)nullptr);
    gemm_qkv_kernel<<<CM / 64, 256, 0, stream>>>(xk, WTk, bk, kb, (__bf16*)nullptr);
    gemm_qkv_kernel<<<CM / 64, 256, 0, stream>>>(xv, WTv, bv, vbuf, vt);

    outl_flash_kernel<<<dim3(CBH, 16), 256, 0, stream>>>(qb, kb, vt, mrow, linv, attnl);
    outr_kernel<<<dim3(CBH, 16), 256, 0, stream>>>(qb, kb, vt, mrow, linv, attnr);

    gemm_proj_kernel<<<CM / 64, 256, 0, stream>>>(attnl, WTp, bp, yl);
    gemm_proj_kernel<<<CM / 64, 256, 0, stream>>>(attnr, WTp, bp, yr);

    ln_res_kernel<<<CM, 256, 0, stream>>>(yl, xl, ln_g, ln_b, outF);
    ln_res_kernel<<<CM, 256, 0, stream>>>(yr, xr, rn_g, rn_b, outF + (size_t)CM * CE);
}
